// DiffMultiHeadAttention_71296457113972
// MI455X (gfx1250) — matrix operand order repaired and checked by reading, not compiled, not run on hardware
//
#include <hip/hip_runtime.h>
#include <hip/hip_bf16.h>
#include <cstdint>

// ---------------------------------------------------------------------------
// DiffAttention forward for MI455X (gfx1250, wave32, WMMA).
// Precision: bf16 WMMA operands, f32 accumulation, f32 softmax/RMS statistics.
// v3: fixed async-to-LDS builtin signature (addrspace-qualified v4i32 pointers).
// ---------------------------------------------------------------------------

typedef __bf16 bf16;
typedef __attribute__((ext_vector_type(8)))  bf16  v8bf;
typedef __attribute__((ext_vector_type(16))) bf16  v16bf;
typedef __attribute__((ext_vector_type(8)))  float v8f;

#define LAMBDA_INIT            0.7836057665316245f
#define ONE_MINUS_LAMBDA_INIT  0.21639423346837547f

#if defined(__has_builtin)
#if __has_builtin(__builtin_amdgcn_global_load_async_to_lds_b128)
#define ASYNC_G2L 1
#endif
#if __has_builtin(__builtin_amdgcn_s_wait_asynccnt)
#define ASYNC_WAIT_BUILTIN 1
#endif
#endif

#if defined(ASYNC_G2L)
typedef int v4i __attribute__((vector_size(16)));
typedef __attribute__((address_space(1))) v4i* g_v4i_ptr;   // global
typedef __attribute__((address_space(3))) v4i* l_v4i_ptr;   // LDS
#endif

__device__ __forceinline__ bf16 f2bf(float f) {
  unsigned u = __builtin_bit_cast(unsigned, f);
  u += 0x7FFFu + ((u >> 16) & 1u);               // round-to-nearest-even
  unsigned short h = (unsigned short)(u >> 16);
  return __builtin_bit_cast(bf16, h);
}

// Copy 16 bf16 (32B) from global to LDS. Async (ASYNCcnt) when available.
__device__ __forceinline__ void g2l_16bf(const bf16* g, bf16* l) {
#if defined(ASYNC_G2L)
  unsigned long long ga = (unsigned long long)(size_t)g;
  unsigned la = (unsigned)(size_t)l;             // low 32 bits = LDS byte offset
  __builtin_amdgcn_global_load_async_to_lds_b128((g_v4i_ptr)ga,
                                                 (l_v4i_ptr)la, 0, 0);
  __builtin_amdgcn_global_load_async_to_lds_b128((g_v4i_ptr)(ga + 16),
                                                 (l_v4i_ptr)(la + 16u), 0, 0);
#else
  *(v8bf*)l       = *(const v8bf*)g;
  *(v8bf*)(l + 8) = *(const v8bf*)(g + 8);
#endif
}

__device__ __forceinline__ void wait_async0() {
#if defined(ASYNC_G2L)
#if defined(ASYNC_WAIT_BUILTIN)
  __builtin_amdgcn_s_wait_asynccnt(0);
#else
  asm volatile("s_wait_asynccnt 0x0" ::: "memory");
#endif
#endif
}

// B-fragment / contiguous-16 load (lane holds 16 consecutive K at fixed n)
__device__ __forceinline__ v16bf ld_frag16(const bf16* p) {
  v8bf lo = *(const v8bf*)p;
  v8bf hi = *(const v8bf*)(p + 8);
  return __builtin_shufflevector(lo, hi, 0,1,2,3,4,5,6,7,8,9,10,11,12,13,14,15);
}
// A-fragment load: lane holds K {base..base+15}
__device__ __forceinline__ v16bf ld_frag_a(const bf16* p) {
  v8bf lo = *(const v8bf*)p;
  v8bf hi = *(const v8bf*)(p + 8);
  return __builtin_shufflevector(lo, hi, 0,1,2,3,4,5,6,7,8,9,10,11,12,13,14,15);
}

__device__ __forceinline__ v8f wmma_bf16(v16bf a, v16bf b, v8f c) {
  return __builtin_amdgcn_wmma_f32_16x16x32_bf16(false, a, false, b, (short)0, c,
                                                 false, false);
}

// ---------------------------------------------------------------------------
// Streaming f32 -> bf16 conversion (RNE), 8 elements/thread.
// ---------------------------------------------------------------------------
__global__ __launch_bounds__(256)
void cvt_f32_to_bf16(const float* __restrict__ src, bf16* __restrict__ dst, int n)
{
  const int i = (blockIdx.x * 256 + threadIdx.x) * 8;
  if (i + 8 <= n) {
    float buf[8];
    *(float4*)&buf[0] = *(const float4*)(src + i);
    *(float4*)&buf[4] = *(const float4*)(src + i + 4);
    v8bf o;
#pragma unroll
    for (int j = 0; j < 8; ++j) o[j] = f2bf(buf[j]);
    *(v8bf*)(dst + i) = o;
  }
}

// ---------------------------------------------------------------------------
// Pure-bf16 GEMM: out = A @ W^T + bias. A [M,K] bf16, W [N,K] bf16, bias f32.
// Block tile 128x128 (8 waves as 2x4, wave tile 64x32), K-step 32,
// double-buffered LDS staging (async-to-LDS when available).
// MODE 0: f32 row-major out.  MODE 1: bf16 out, head-major [b,16,s,64].
// MODE 2: bf16 out, head-major transposed [b,8,d(128),s] (B-operand of aw@v).
// ---------------------------------------------------------------------------
template <int MODE>
__global__ __launch_bounds__(256)
void gemm_bf16(const bf16* __restrict__ A, const bf16* __restrict__ W,
               const float* __restrict__ bias,
               float* __restrict__ outf, bf16* __restrict__ outb,
               int M, int N, int K)
{
  __shared__ __align__(16) bf16 As[2][128 * 40];   // stride 40 (80B) pads banks
  __shared__ __align__(16) bf16 Bs[2][128 * 48];   // stride 48 (96B)

  const int tid  = threadIdx.x;
  const int wave = tid >> 5, lane = tid & 31;
  const int l16  = lane & 15, lh = lane >> 4;
  const int waveM = wave >> 2, waveN = wave & 3;
  const int rowBase = blockIdx.y * 128, colBase = blockIdx.x * 128;

  v8f zerov = {0.f,0.f,0.f,0.f,0.f,0.f,0.f,0.f};
  v8f c[4][2];
#pragma unroll
  for (int mt = 0; mt < 4; ++mt)
#pragma unroll
    for (int nt = 0; nt < 2; ++nt) c[mt][nt] = zerov;

  const int r = tid & 127, seg = tid >> 7, kc = seg * 16;
  const bf16* aptr = A + (size_t)(rowBase + r) * K + kc;
  const bf16* wptr = W + (size_t)(colBase + r) * K + kc;

  // prologue: stage tile 0 into buffer 0
  g2l_16bf(aptr, &As[0][r * 40 + kc]);
  g2l_16bf(wptr, &Bs[0][r * 48 + kc]);
  wait_async0();
  __syncthreads();

  const int NT = K / 32;
  for (int kt = 0; kt < NT; ++kt) {
    const int cur = kt & 1, nxt = cur ^ 1;
    if (kt + 1 < NT) {            // stage next tile while computing current
      g2l_16bf(aptr + (kt + 1) * 32, &As[nxt][r * 40 + kc]);
      g2l_16bf(wptr + (kt + 1) * 32, &Bs[nxt][r * 48 + kc]);
    }
    if (kt + 2 < NT) {
      __builtin_prefetch(aptr + (kt + 2) * 32, 0, 1);   // global_prefetch_b8
      __builtin_prefetch(wptr + (kt + 2) * 32, 0, 1);
    }

    v16bf afr[4], bfr[2];
#pragma unroll
    for (int mt = 0; mt < 4; ++mt)
      afr[mt] = ld_frag_a(&As[cur][(waveM * 64 + mt * 16 + l16) * 40 + lh * 16]);
#pragma unroll
    for (int nt = 0; nt < 2; ++nt)
      bfr[nt] = ld_frag16(&Bs[cur][(waveN * 32 + nt * 16 + l16) * 48 + lh * 16]);
#pragma unroll
    for (int mt = 0; mt < 4; ++mt)
#pragma unroll
      for (int nt = 0; nt < 2; ++nt)
        c[mt][nt] = wmma_bf16(afr[mt], bfr[nt], c[mt][nt]);

    if (kt + 1 < NT) {
      wait_async0();
      __syncthreads();
    }
  }

  // ---- epilogue ----
#pragma unroll
  for (int mt = 0; mt < 4; ++mt)
#pragma unroll
    for (int nt = 0; nt < 2; ++nt) {
      const int col  = colBase + waveN * 32 + nt * 16 + l16;
      const float bc = bias[col];
#pragma unroll
      for (int rr = 0; rr < 8; ++rr) {
        const int row   = rowBase + waveM * 64 + mt * 16 + rr + 8 * lh;
        const float val = c[mt][nt][rr] + bc;
        if constexpr (MODE == 0) {
          outf[(size_t)row * N + col] = val;
        } else if constexpr (MODE == 1) {    // q/k: [b,16,s,64]
          const int bb = row >> 11, s = row & 2047;
          const int head = col >> 6, d = col & 63;
          outb[(((size_t)(bb * 16 + head)) * 2048 + s) * 64 + d] = f2bf(val);
        } else {                             // v: [b,8,d(128),s] transposed
          const int bb = row >> 11, s = row & 2047;
          const int head = col >> 7, d = col & 127;
          outb[(((size_t)(bb * 8 + head)) * 128 + d) * 2048 + s] = f2bf(val);
        }
      }
    }
}

// ---------------------------------------------------------------------------
// Scores + softmax + differential combine.
// Block = (b, head-pair hp, 64 q-rows). Wave w owns key columns [w*256,w*256+256).
// Per head: P1 WMMA scores -> global (f32) + running row-max; P2 exp/row-sum;
// P3 normalize (even head, in aw region) / combine (odd): aw = pe - lam*po,
// written f32 to aw output and bf16 to workspace for the aw@v GEMM.
// ---------------------------------------------------------------------------
__global__ __launch_bounds__(256)
void diffattn_scores(const bf16* __restrict__ qbf, const bf16* __restrict__ kbf,
                     float* __restrict__ aw, float* __restrict__ oddb,
                     bf16* __restrict__ awbf,
                     const float* __restrict__ lq1, const float* __restrict__ lk1,
                     const float* __restrict__ lq2, const float* __restrict__ lk2)
{
  const int tid  = threadIdx.x;
  const int wave = tid >> 5, lane = tid & 31;
  const int l16  = lane & 15, lh = lane >> 4;
  const int rowBase = blockIdx.x * 64;
  const int b = blockIdx.y >> 3, hp = blockIdx.y & 7;

  __shared__ float wstat[8][64];
  __shared__ float rowred[64];
  __shared__ float rowinv[64];

  // lambda scalar (64-element dot products, f32)
  float s1 = 0.f, s2 = 0.f;
  for (int i = 0; i < 64; ++i) { s1 += lq1[i] * lk1[i]; s2 += lq2[i] * lk2[i]; }
  const float lam = __expf(s1) - __expf(s2) + LAMBDA_INIT;

  const size_t stripe = ((size_t)(b * 8 + hp) * 2048 + rowBase) * 2048;

  for (int which = 0; which < 2; ++which) {
    const int head = hp * 2 + which;
    const bf16* qh = qbf + ((size_t)(b * 16 + head) * 2048 + rowBase) * 64;
    const bf16* kh = kbf + (size_t)(b * 16 + head) * 2048 * 64;
    float* dstp = (which == 0 ? aw : oddb) + stripe;

    // q fragments for 64 rows x K=64, held in registers for whole stripe
    v16bf aq[4][2];
#pragma unroll
    for (int mt = 0; mt < 4; ++mt)
#pragma unroll
      for (int ks = 0; ks < 2; ++ks)
        aq[mt][ks] = ld_frag_a(qh + (size_t)(mt * 16 + l16) * 64 + ks * 32 + lh * 16);

    float vmax[4][8];
#pragma unroll
    for (int mt = 0; mt < 4; ++mt)
#pragma unroll
      for (int rr = 0; rr < 8; ++rr) vmax[mt][rr] = -3.0e38f;

    // ---- P1: scores via WMMA ----
    for (int i = 0; i < 16; ++i) {
      const int key0 = wave * 256 + i * 16;
      v16bf bk0 = ld_frag16(kh + (size_t)(key0 + l16) * 64 + 0  + lh * 16);
      v16bf bk1 = ld_frag16(kh + (size_t)(key0 + l16) * 64 + 32 + lh * 16);
#pragma unroll
      for (int mt = 0; mt < 4; ++mt) {
        v8f cc = {0.f,0.f,0.f,0.f,0.f,0.f,0.f,0.f};
        cc = wmma_bf16(aq[mt][0], bk0, cc);
        cc = wmma_bf16(aq[mt][1], bk1, cc);
#pragma unroll
        for (int rr = 0; rr < 8; ++rr) {
          const float sv = cc[rr] * 0.125f;     // 1/sqrt(64)
          vmax[mt][rr] = fmaxf(vmax[mt][rr], sv);
          dstp[(size_t)(mt * 16 + rr + 8 * lh) * 2048 + key0 + l16] = sv;
        }
      }
    }
    asm volatile("s_wait_storecnt 0x0" ::: "memory");

    // row max: reduce across the 16 lanes sharing a row, then across waves
#pragma unroll
    for (int mt = 0; mt < 4; ++mt)
#pragma unroll
      for (int rr = 0; rr < 8; ++rr) {
        float m = vmax[mt][rr];
        m = fmaxf(m, __shfl_xor(m, 1, 32));
        m = fmaxf(m, __shfl_xor(m, 2, 32));
        m = fmaxf(m, __shfl_xor(m, 4, 32));
        m = fmaxf(m, __shfl_xor(m, 8, 32));
        if (l16 == 0) wstat[wave][mt * 16 + rr + 8 * lh] = m;
      }
    __syncthreads();
    if (tid < 64) {
      float m = wstat[0][tid];
#pragma unroll
      for (int w = 1; w < 8; ++w) m = fmaxf(m, wstat[w][tid]);
      rowred[tid] = m;
    }
    __syncthreads();

    // ---- P2: exp(x - rowmax), row sums (same wave re-reads its own columns) --
    for (int rr = 0; rr < 64; ++rr) {
      const float rm = rowred[rr];
      float s = 0.f;
#pragma unroll
      for (int j = 0; j < 8; ++j) {
        const size_t idx = (size_t)rr * 2048 + wave * 256 + j * 32 + lane;
        const float e = __expf(dstp[idx] - rm);
        dstp[idx] = e;
        s += e;
      }
      s += __shfl_xor(s, 1, 32);  s += __shfl_xor(s, 2, 32);
      s += __shfl_xor(s, 4, 32);  s += __shfl_xor(s, 8, 32);
      s += __shfl_xor(s, 16, 32);
      if (lane == 0) wstat[wave][rr] = s;
    }
    asm volatile("s_wait_storecnt 0x0" ::: "memory");
    __syncthreads();
    if (tid < 64) {
      float s = 0.f;
#pragma unroll
      for (int w = 0; w < 8; ++w) s += wstat[w][tid];
      rowinv[tid] = 1.0f / s;
    }
    __syncthreads();

    // ---- P3: normalize (even) / combine (odd) ----
    if (which == 0) {
      for (int rr = 0; rr < 64; ++rr) {
        const float inv = rowinv[rr];
#pragma unroll
        for (int j = 0; j < 8; ++j) {
          const size_t idx = (size_t)rr * 2048 + wave * 256 + j * 32 + lane;
          dstp[idx] *= inv;
        }
      }
      asm volatile("s_wait_storecnt 0x0" ::: "memory");
    } else {
      float* evp = aw + stripe;
      for (int rr = 0; rr < 64; ++rr) {
        const float inv = rowinv[rr];
#pragma unroll
        for (int j = 0; j < 8; ++j) {
          const size_t idx = (size_t)rr * 2048 + wave * 256 + j * 32 + lane;
          const float po  = dstp[idx] * inv;
          const float fin = evp[idx] - lam * po;
          evp[idx] = fin;                    // final aw (f32 output)
          awbf[stripe + idx] = f2bf(fin);    // bf16 copy for aw@v GEMM
        }
      }
    }
    __syncthreads();   // protect wstat reuse for the second head
  }
}

// ---------------------------------------------------------------------------
// attn = aw @ v, fused RMS-norm over 2dk=128 + subln_w + (1-lambda_init).
// Block = (b,h, 64 rows) x all 128 cols; wave w owns d-tile w. K=2048, step 32.
// v is pre-transposed [b,h,d,s] so B-fragments are contiguous 32B loads.
// Double-buffered LDS stripe of aw_bf (async staging when available).
// Output attn_bf is [b, s, h*128] bf16 for the final projection GEMM.
// ---------------------------------------------------------------------------
__global__ __launch_bounds__(256)
void attn_av(const bf16* __restrict__ awbf, const bf16* __restrict__ vbf,
             const float* __restrict__ subw, bf16* __restrict__ attnb)
{
  __shared__ __align__(16) bf16 As[2][64 * 40];
  __shared__ float wss[8][64];
  __shared__ float rinv[64];

  const int tid  = threadIdx.x;
  const int wave = tid >> 5, lane = tid & 31;
  const int l16  = lane & 15, lh = lane >> 4;
  const int rowBase = blockIdx.x * 64;
  const int bh = blockIdx.y, b = bh >> 3, h = bh & 7;

  const bf16* arow = awbf + ((size_t)bh * 2048 + rowBase) * 2048;
  const bf16* vh   = vbf + (size_t)bh * 128 * 2048;
  const int d = wave * 16 + l16;

  v8f zerov = {0.f,0.f,0.f,0.f,0.f,0.f,0.f,0.f};
  v8f c[4];
#pragma unroll
  for (int mt = 0; mt < 4; ++mt) c[mt] = zerov;

  const int sr = tid & 63, seg = tid >> 6;    // A staging: 64 rows x 4 K-segments
  const bf16* astage = arow + (size_t)sr * 2048 + seg * 8;

  g2l_16bf(astage, &As[0][sr * 40 + seg * 8]);
  wait_async0();
  __syncthreads();

  for (int kt = 0; kt < 64; ++kt) {
    const int cur = kt & 1, nxt = cur ^ 1;
    if (kt + 1 < 64)
      g2l_16bf(astage + (kt + 1) * 32, &As[nxt][sr * 40 + seg * 8]);

    v16bf bv = ld_frag16(vh + (size_t)d * 2048 + kt * 32 + lh * 16);
#pragma unroll
    for (int mt = 0; mt < 4; ++mt) {
      v16bf av = ld_frag_a(&As[cur][(mt * 16 + l16) * 40 + lh * 16]);
      c[mt] = wmma_bf16(av, bv, c[mt]);
    }

    if (kt + 1 < 64) {
      wait_async0();
      __syncthreads();
    }
  }

  // RMS statistics: sum of squares over the 128 d-columns of each row
#pragma unroll
  for (int mt = 0; mt < 4; ++mt)
#pragma unroll
    for (int rr = 0; rr < 8; ++rr) {
      float ss = c[mt][rr] * c[mt][rr];
      ss += __shfl_xor(ss, 1, 32);  ss += __shfl_xor(ss, 2, 32);
      ss += __shfl_xor(ss, 4, 32);  ss += __shfl_xor(ss, 8, 32);
      if (l16 == 0) wss[wave][mt * 16 + rr + 8 * lh] = ss;
    }
  __syncthreads();
  if (tid < 64) {
    float t = 0.f;
#pragma unroll
    for (int w = 0; w < 8; ++w) t += wss[w][tid];
    rinv[tid] = rsqrtf(t * (1.0f / 128.0f) + 1e-5f);
  }
  __syncthreads();

  const float swd = subw[d] * ONE_MINUS_LAMBDA_INIT;
#pragma unroll
  for (int mt = 0; mt < 4; ++mt)
#pragma unroll
    for (int rr = 0; rr < 8; ++rr) {
      const int rl = mt * 16 + rr + 8 * lh;
      const float val = c[mt][rr] * rinv[rl] * swd;
      const int s = rowBase + rl;
      attnb[((size_t)(b * 2048 + s)) * 1024 + h * 128 + d] = f2bf(val);
    }
}

// ---------------------------------------------------------------------------
// Host launcher
// ---------------------------------------------------------------------------
extern "C" void kernel_launch(void* const* d_in, const int* in_sizes, int n_in,
                              void* d_out, int out_size, void* d_ws, size_t ws_size,
                              hipStream_t stream) {
  (void)in_sizes; (void)n_in; (void)out_size; (void)ws_size;

  const float* query = (const float*)d_in[0];
  const float* key_  = (const float*)d_in[1];
  const float* value = (const float*)d_in[2];
  const float* wq_w  = (const float*)d_in[3];
  const float* wq_b  = (const float*)d_in[4];
  const float* wk_w  = (const float*)d_in[5];
  const float* wk_b  = (const float*)d_in[6];
  const float* wv_w  = (const float*)d_in[7];
  const float* wv_b  = (const float*)d_in[8];
  const float* fc_w  = (const float*)d_in[9];
  const float* fc_b  = (const float*)d_in[10];
  const float* lq1   = (const float*)d_in[11];
  const float* lk1   = (const float*)d_in[12];
  const float* lq2   = (const float*)d_in[13];
  const float* lk2   = (const float*)d_in[14];
  const float* subw  = (const float*)d_in[15];

  float* out = (float*)d_out;
  float* aw  = out + (size_t)2 * 2048 * 1024;           // aw follows out

  // workspace layout (bytes):
  //  q_bf 8M | k_bf 8M | v_bf 8M | aw_bf 128M | attn_bf 8M | odd f32 256M |
  //  qx 8M | kx 8M | vx 8M | wq 2M | wk 2M | wv 2M | fc 2M      (~448 MB)
  char* ws = (char*)d_ws;
  bf16* q_bf    = (bf16*)(ws);
  bf16* k_bf    = (bf16*)(ws + 8388608);
  bf16* v_bf    = (bf16*)(ws + 16777216);
  bf16* aw_bf   = (bf16*)(ws + 25165824);
  bf16* attn_bf = (bf16*)(ws + 159383552);
  float* odd_ws = (float*)(ws + 167772160);
  bf16* qx_bf   = (bf16*)(ws + 436207616);
  bf16* kx_bf   = (bf16*)(ws + 444596224);
  bf16* vx_bf   = (bf16*)(ws + 452984832);
  bf16* wq_bf   = (bf16*)(ws + 461373440);
  bf16* wk_bf   = (bf16*)(ws + 463470592);
  bf16* wv_bf   = (bf16*)(ws + 465567744);
  bf16* fc_bf   = (bf16*)(ws + 467664896);

  const int M = 4096, N = 1024, K = 1024;
  const int nAct = M * K;      // 4,194,304
  const int nWgt = N * K;      // 1,048,576
  dim3 bb(256);

  // one-shot f32 -> bf16 conversion (each tensor read exactly once)
  cvt_f32_to_bf16<<<dim3(nAct / 2048), bb, 0, stream>>>(query, qx_bf, nAct);
  cvt_f32_to_bf16<<<dim3(nAct / 2048), bb, 0, stream>>>(key_,  kx_bf, nAct);
  cvt_f32_to_bf16<<<dim3(nAct / 2048), bb, 0, stream>>>(value, vx_bf, nAct);
  cvt_f32_to_bf16<<<dim3(nWgt / 2048), bb, 0, stream>>>(wq_w, wq_bf, nWgt);
  cvt_f32_to_bf16<<<dim3(nWgt / 2048), bb, 0, stream>>>(wk_w, wk_bf, nWgt);
  cvt_f32_to_bf16<<<dim3(nWgt / 2048), bb, 0, stream>>>(wv_w, wv_bf, nWgt);
  cvt_f32_to_bf16<<<dim3(nWgt / 2048), bb, 0, stream>>>(fc_w, fc_bf, nWgt);

  dim3 gg(N / 128, M / 128);
  gemm_bf16<1><<<gg, bb, 0, stream>>>(qx_bf, wq_bf, wq_b, nullptr, q_bf, M, N, K);
  gemm_bf16<1><<<gg, bb, 0, stream>>>(kx_bf, wk_bf, wk_b, nullptr, k_bf, M, N, K);
  gemm_bf16<2><<<gg, bb, 0, stream>>>(vx_bf, wv_bf, wv_b, nullptr, v_bf, M, N, K);

  diffattn_scores<<<dim3(32, 16), bb, 0, stream>>>(q_bf, k_bf, aw, odd_ws, aw_bf,
                                                   lq1, lk1, lq2, lk2);

  attn_av<<<dim3(32, 16), bb, 0, stream>>>(aw_bf, v_bf, subw, attn_bf);

  gemm_bf16<0><<<gg, bb, 0, stream>>>(attn_bf, fc_bf, fc_b, out, nullptr, M, N, K);
}
